// EnhancedHybridModel_60679297958333
// MI455X (gfx1250) — matrix operand order repaired and checked by reading, not compiled, not run on hardware
//
#include <hip/hip_runtime.h>
#include <hip/hip_bf16.h>
#include <math.h>

// EnhancedHybridModel on gfx1250:
//  - fixed circuit (3 layers RX/RY/RZ + CNOT ladder) precomputed as 256x256
//    complex unitary, stored as f16 WMMA-B fragments in d_ws
//  - batch pipeline: MLP front (LDS cooperative) -> real product state ->
//    complex GEMM via v_wmma_f32_16x16x32_f16 -> probs -> Z signs -> MLP tail

#define EPS 1e-5f

typedef _Float16 v16h __attribute__((ext_vector_type(16)));
typedef _Float16 v8h  __attribute__((ext_vector_type(8)));
typedef float    v8f  __attribute__((ext_vector_type(8)));

// ---------------------------------------------------------------------------
// Kernel A: build the fixed-circuit unitary (transpose-applied, matching the
// reference einsum) by simulating 256 basis states; write f16 re/im in
// swizzled WMMA-B fragment layout:  off = ((nt*8 + kt)*32 + lane)*16 + e
// with lane = 16*(klocal/16) + (n%16), e = klocal%16, kt = k/32, nt = n/16.
// ---------------------------------------------------------------------------
__global__ __launch_bounds__(256) void build_unitary_kernel(
    const float* __restrict__ qw,      // [3][8][3]
    _Float16* __restrict__ Bre,        // 256*256 swizzled
    _Float16* __restrict__ Bim)
{
  __shared__ float2 st[32 * 256];      // 32 basis columns x 256 amps = 64KB
  const int t  = threadIdx.x;
  const int bc = blockIdx.x;           // owns basis states [bc*32, bc*32+32)

  // init: column lc holds basis state k = bc*32 + lc
  for (int v = 0; v < 32; ++v) {
    int lc = v;                        // 32 cols, 256 amps each; thread t = amp n
    int n  = t;
    float2 z; z.x = (n == bc * 32 + lc) ? 1.f : 0.f; z.y = 0.f;
    st[lc * 256 + n] = z;
  }
  __syncthreads();

  for (int l = 0; l < 3; ++l) {
    for (int q = 0; q < 8; ++q) {
      const int bp = 7 - q;
      for (int gidx = 0; gidx < 3; ++gidx) {
        float th = qw[(l * 8 + q) * 3 + gidx];
        float c = cosf(0.5f * th), s = sinf(0.5f * th);
        float2 u00, u01, u10, u11;
        if (gidx == 0) {                       // RX (symmetric, transpose == same)
          u00 = make_float2(c, 0.f); u01 = make_float2(0.f, -s);
          u10 = make_float2(0.f, -s); u11 = make_float2(c, 0.f);
        } else if (gidx == 1) {                // RY, reference applies U^T
          u00 = make_float2(c, 0.f); u01 = make_float2(s, 0.f);
          u10 = make_float2(-s, 0.f); u11 = make_float2(c, 0.f);
        } else {                               // RZ (diagonal)
          u00 = make_float2(c, -s); u01 = make_float2(0.f, 0.f);
          u10 = make_float2(0.f, 0.f); u11 = make_float2(c, s);
        }
        for (int v = 0; v < 16; ++v) {         // 32 cols * 128 pairs = 4096 tasks
          int tid = v * 256 + t;
          int col = tid >> 7;
          int pr  = tid & 127;
          int low = pr & ((1 << bp) - 1);
          int high = pr >> bp;
          int i0 = (high << (bp + 1)) | low;
          int i1 = i0 | (1 << bp);
          float2 a = st[col * 256 + i0], b = st[col * 256 + i1];
          float2 r0, r1;
          r0.x = u00.x * a.x - u00.y * a.y + u01.x * b.x - u01.y * b.y;
          r0.y = u00.x * a.y + u00.y * a.x + u01.x * b.y + u01.y * b.x;
          r1.x = u10.x * a.x - u10.y * a.y + u11.x * b.x - u11.y * b.y;
          r1.y = u10.x * a.y + u10.y * a.x + u11.x * b.y + u11.y * b.x;
          st[col * 256 + i0] = r0;
          st[col * 256 + i1] = r1;
        }
        __syncthreads();
      }
    }
    // CNOT ladder: swap pairs (idx, idx^tgt) where ctrl bit == 1, tgt bit == 0
    for (int i = 0; i < 8; ++i) {
      for (int j = i + 1; j < 8; ++j) {
        const int bctrl = 7 - i, btgt = 7 - j;   // bctrl > btgt
        for (int v = 0; v < 8; ++v) {            // 32 cols * 64 swaps = 2048 tasks
          int tid = v * 256 + t;
          int col = tid >> 6;
          int sw  = tid & 63;
          int low  = sw & ((1 << btgt) - 1);
          int mid  = (sw >> btgt) & ((1 << (bctrl - btgt - 1)) - 1);
          int high = sw >> (bctrl - 1);
          int idx = (high << (bctrl + 1)) | (1 << bctrl) | (mid << (btgt + 1)) | low;
          int idx2 = idx ^ (1 << btgt);
          float2 a = st[col * 256 + idx], b = st[col * 256 + idx2];
          st[col * 256 + idx]  = b;
          st[col * 256 + idx2] = a;
        }
        __syncthreads();
      }
    }
  }

  // write out B[k][n] = amp_n(basis k) into swizzled fragment layout
  for (int v = 0; v < 32; ++v) {
    int tid = v * 256 + t;
    int lc = tid >> 8;
    int n  = tid & 255;
    float2 a = st[lc * 256 + n];
    int g = lc >> 4, e = lc & 15;
    int nt = n >> 4, nl = n & 15;
    int lane = g * 16 + nl;
    int off = ((nt * 8 + bc) * 32 + lane) * 16 + e;
    Bre[off] = (_Float16)a.x;
    Bim[off] = (_Float16)a.y;
  }
}

// ---------------------------------------------------------------------------
// Kernel B: 64 batch rows per block (4 waves; each wave owns 16 rows).
// ---------------------------------------------------------------------------
__global__ __launch_bounds__(128) void hybrid_forward_kernel(
    const float* __restrict__ x,
    const float* __restrict__ W1, const float* __restrict__ b1,
    const float* __restrict__ g1, const float* __restrict__ bt1,
    const float* __restrict__ m1, const float* __restrict__ v1,
    const float* __restrict__ W2, const float* __restrict__ b2,
    const float* __restrict__ g2, const float* __restrict__ bt2,
    const float* __restrict__ m2, const float* __restrict__ v2,
    const float* __restrict__ W3, const float* __restrict__ b3,
    const float* __restrict__ W4, const float* __restrict__ b4,
    const float* __restrict__ g4, const float* __restrict__ bt4,
    const float* __restrict__ m4, const float* __restrict__ v4,
    const float* __restrict__ W5, const float* __restrict__ b5,
    const float* __restrict__ W6, const float* __restrict__ b6,
    const _Float16* __restrict__ Bre, const _Float16* __restrict__ Bim,
    float* __restrict__ out)
{
  __shared__ float    xs[64 * 16];      // 4KB
  __shared__ float    h1s[64 * 128];    // 32KB
  __shared__ float    h2s[64 * 64];     // 16KB
  __shared__ float2   ctab[64 * 8];     // 4KB  (cos, -sin) per row/qubit
  __shared__ _Float16 stateI[64 * 256]; // 32KB f16 product states
  __shared__ float    qbuf[64 * 8];     // 2KB  Z expectations

  const int t    = threadIdx.x;
  const int row0 = blockIdx.x * 64;

  // ---- stage x -----------------------------------------------------------
  for (int v = 0; v < 8; ++v) {
    int idx = v * 128 + t;
    xs[idx] = x[row0 * 16 + idx];
  }
  __syncthreads();

  // ---- layer 1: 16 -> 128, BN, ReLU (thread t owns unit t) ---------------
  {
    float w[16];
    #pragma unroll
    for (int k = 0; k < 16; ++k) w[k] = W1[t * 16 + k];
    float bb = b1[t];
    float sc = g1[t] * rsqrtf(v1[t] + EPS);
    float sh = bt1[t] - m1[t] * sc;
    for (int r = 0; r < 64; ++r) {
      float acc = bb;
      #pragma unroll
      for (int k = 0; k < 16; ++k) acc += xs[r * 16 + k] * w[k];
      h1s[r * 128 + t] = fmaxf(acc * sc + sh, 0.f);
    }
  }
  __syncthreads();

  // ---- layer 2: 128 -> 64, BN, ReLU (thread = (unit, half-of-rows)) ------
  {
    int u = t & 63, hs = t >> 6;
    float acc[32];
    #pragma unroll
    for (int r = 0; r < 32; ++r) acc[r] = 0.f;
    for (int ko = 0; ko < 8; ++ko) {
      float w[16];
      #pragma unroll
      for (int k = 0; k < 16; ++k) w[k] = W2[u * 128 + ko * 16 + k];
      #pragma unroll
      for (int r = 0; r < 32; ++r) {
        float a = 0.f;
        #pragma unroll
        for (int k = 0; k < 16; ++k)
          a += h1s[(hs * 32 + r) * 128 + ko * 16 + k] * w[k];
        acc[r] += a;
      }
    }
    float bb = b2[u];
    float sc = g2[u] * rsqrtf(v2[u] + EPS);
    float sh = bt2[u] - m2[u] * sc;
    #pragma unroll
    for (int r = 0; r < 32; ++r)
      h2s[(hs * 32 + r) * 64 + u] = fmaxf((acc[r] + bb) * sc + sh, 0.f);
  }
  __syncthreads();

  // ---- layer 3: 64 -> 8, tanh -> angle table (cos, -sin of theta/2) ------
  {
    int u = t & 7, rg = t >> 3;
    for (int rr = 0; rr < 4; ++rr) {
      int r = rg * 4 + rr;
      float acc = b3[u];
      #pragma unroll
      for (int k = 0; k < 64; ++k) acc += h2s[r * 64 + k] * W3[u * 64 + k];
      float pre = tanhf(acc);
      // reference einsum applies RY^T: |0> -> (cos, -sin)
      ctab[r * 8 + u] = make_float2(cosf(0.5f * pre), -sinf(0.5f * pre));
    }
  }
  __syncthreads();

  // ---- product state (real), f16, laid out [row][k] ----------------------
  for (int v = 0; v < 128; ++v) {
    int idx = v * 128 + t;
    int r = idx >> 8, k = idx & 255;
    float p = 1.f;
    #pragma unroll
    for (int i = 0; i < 8; ++i) {
      float2 cs = ctab[r * 8 + i];
      p *= ((k >> (7 - i)) & 1) ? cs.y : cs.x;
    }
    stateI[idx] = (_Float16)p;      // idx == r*256 + k
  }
  __syncthreads();

  // ---- complex GEMM: amp[16x256] = state[16x256] @ U^T, via f16 WMMA -----
  const int lane = t & 31, w = t >> 5;
  const int g = lane >> 4, m = lane & 15;
  const int lrow = w * 16 + m;

  // A panel: contiguous k order, same as B packing -> lane holds K = 16g + {0..15}
  v16h apan[8];
  #pragma unroll
  for (int kt = 0; kt < 8; ++kt) {
    union { v16h v; v8h h[2]; } u;
    u.h[0] = *(const v8h*)&stateI[lrow * 256 + kt * 32 + 16 * g];
    u.h[1] = *(const v8h*)&stateI[lrow * 256 + kt * 32 + 16 * g + 8];
    apan[kt] = u.v;
  }

  float qacc[8][8];
  #pragma unroll
  for (int r = 0; r < 8; ++r)
    #pragma unroll
    for (int i = 0; i < 8; ++i) qacc[r][i] = 0.f;

  for (int nt = 0; nt < 16; ++nt) {
    v8f ar = {0.f, 0.f, 0.f, 0.f, 0.f, 0.f, 0.f, 0.f};
    v8f ai = {0.f, 0.f, 0.f, 0.f, 0.f, 0.f, 0.f, 0.f};
    #pragma unroll
    for (int kt = 0; kt < 8; ++kt) {
      v16h brf = *(const v16h*)(Bre + ((nt * 8 + kt) * 32 + lane) * 16);
      v16h bif = *(const v16h*)(Bim + ((nt * 8 + kt) * 32 + lane) * 16);
      ar = __builtin_amdgcn_wmma_f32_16x16x32_f16(
          false, apan[kt], false, brf, (short)0, ar, false, false);
      ai = __builtin_amdgcn_wmma_f32_16x16x32_f16(
          false, apan[kt], false, bif, (short)0, ai, false, false);
    }
    // probs and signed accumulation into Z expectations
    int N = nt * 16 + m;
    float sg[8];
    #pragma unroll
    for (int i = 0; i < 8; ++i) sg[i] = ((N >> (7 - i)) & 1) ? -1.f : 1.f;
    #pragma unroll
    for (int r = 0; r < 8; ++r) {
      float p = ar[r] * ar[r] + ai[r] * ai[r];
      #pragma unroll
      for (int i = 0; i < 8; ++i) qacc[r][i] += p * sg[i];
    }
  }

  // reduce over the 16 lanes of each half-wave (they cover distinct N)
  #pragma unroll
  for (int r = 0; r < 8; ++r)
    #pragma unroll
    for (int i = 0; i < 8; ++i) {
      float vsum = qacc[r][i];
      vsum += __shfl_xor(vsum, 1);
      vsum += __shfl_xor(vsum, 2);
      vsum += __shfl_xor(vsum, 4);
      vsum += __shfl_xor(vsum, 8);
      qacc[r][i] = vsum;
    }
  if (m == 0) {
    #pragma unroll
    for (int r = 0; r < 8; ++r)
      #pragma unroll
      for (int i = 0; i < 8; ++i)
        qbuf[(w * 16 + g * 8 + r) * 8 + i] = qacc[r][i];
  }
  __syncthreads();

  // ---- tail MLP: 8 -> 32 (BN,ReLU) -> 16 (ReLU) -> 1 ---------------------
  if (t < 64) {
    float q[8];
    #pragma unroll
    for (int i = 0; i < 8; ++i) q[i] = qbuf[t * 8 + i];
    float h4[32];
    #pragma unroll
    for (int o = 0; o < 32; ++o) {
      float acc = b4[o];
      #pragma unroll
      for (int i = 0; i < 8; ++i) acc += q[i] * W4[o * 8 + i];
      float sc = g4[o] * rsqrtf(v4[o] + EPS);
      h4[o] = fmaxf((acc - m4[o]) * sc + bt4[o], 0.f);
    }
    float h5[16];
    #pragma unroll
    for (int o = 0; o < 16; ++o) {
      float acc = b5[o];
      #pragma unroll
      for (int k = 0; k < 32; ++k) acc += h4[k] * W5[o * 32 + k];
      h5[o] = fmaxf(acc, 0.f);
    }
    float acc = b6[0];
    #pragma unroll
    for (int k = 0; k < 16; ++k) acc += h5[k] * W6[k];
    out[row0 + t] = acc;
  }
}

// ---------------------------------------------------------------------------
extern "C" void kernel_launch(void* const* d_in, const int* in_sizes, int n_in,
                              void* d_out, int out_size, void* d_ws, size_t ws_size,
                              hipStream_t stream) {
  (void)in_sizes; (void)n_in; (void)ws_size;
  const float* x   = (const float*)d_in[0];
  const float* W1  = (const float*)d_in[1];
  const float* b1  = (const float*)d_in[2];
  const float* g1  = (const float*)d_in[3];
  const float* bt1 = (const float*)d_in[4];
  const float* m1  = (const float*)d_in[5];
  const float* v1  = (const float*)d_in[6];
  const float* W2  = (const float*)d_in[7];
  const float* b2  = (const float*)d_in[8];
  const float* g2  = (const float*)d_in[9];
  const float* bt2 = (const float*)d_in[10];
  const float* m2  = (const float*)d_in[11];
  const float* v2  = (const float*)d_in[12];
  const float* W3  = (const float*)d_in[13];
  const float* b3  = (const float*)d_in[14];
  const float* qw  = (const float*)d_in[15];
  const float* W4  = (const float*)d_in[16];
  const float* b4  = (const float*)d_in[17];
  const float* g4  = (const float*)d_in[18];
  const float* bt4 = (const float*)d_in[19];
  const float* m4  = (const float*)d_in[20];
  const float* v4  = (const float*)d_in[21];
  const float* W5  = (const float*)d_in[22];
  const float* b5  = (const float*)d_in[23];
  const float* W6  = (const float*)d_in[24];
  const float* b6  = (const float*)d_in[25];

  _Float16* Bre = (_Float16*)d_ws;          // 128KB
  _Float16* Bim = Bre + 256 * 256;          // 128KB

  build_unitary_kernel<<<8, 256, 0, stream>>>(qw, Bre, Bim);

  hybrid_forward_kernel<<<65536 / 64, 128, 0, stream>>>(
      x, W1, b1, g1, bt1, m1, v1,
      W2, b2, g2, bt2, m2, v2,
      W3, b3,
      W4, b4, g4, bt4, m4, v4,
      W5, b5, W6, b6,
      Bre, Bim, (float*)d_out);
}
